// FlashInfer_KeyFormer_31731218382933
// MI455X (gfx1250) — compile-verified, matrix operand order checked by reading, not run on hardware
//
#include <hip/hip_runtime.h>

typedef __attribute__((ext_vector_type(16))) __bf16 v16bf;
typedef __attribute__((ext_vector_type(8)))  __bf16 v8bf;
typedef __attribute__((ext_vector_type(4)))  __bf16 v4bf;
typedef __attribute__((ext_vector_type(2)))  __bf16 v2bf;
typedef __attribute__((ext_vector_type(8)))  float  v8f;
typedef __attribute__((ext_vector_type(4)))  float  v4f;
typedef __attribute__((ext_vector_type(4)))  unsigned int u32x4;
typedef __attribute__((ext_vector_type(8)))  int i32x8;
typedef __attribute__((ext_vector_type(4)))  int i32x4;

#define M_  256
#define N_  4096
#define D_  128
#define P_  4096
#define H_  32
#define T_  4352            // P + M
#define LDSTR 40            // LDS row stride (bf16) for 32-wide K tiles: 64B + 16B pad
#define KSTR  136           // LDS row stride (bf16) for 128-wide K tiles: 256B + 16B pad

#if __has_builtin(__builtin_amdgcn_tensor_load_to_lds) && \
    __has_builtin(__builtin_amdgcn_s_wait_tensorcnt)
#define USE_TDM 1
#else
#define USE_TDM 0
#endif

static __device__ __forceinline__ v16bf cat16(v8bf lo, v8bf hi) {
  return __builtin_shufflevector(lo, hi, 0,1,2,3,4,5,6,7,8,9,10,11,12,13,14,15);
}
static __device__ __forceinline__ v8f wmma_bf16(v16bf a, v16bf b, v8f c) {
  return __builtin_amdgcn_wmma_f32_16x16x32_bf16(false, a, false, b, (short)0, c, false, false);
}

#if USE_TDM
// LDS offset of a __shared__ object (generic -> addrspace(3) -> int)
static __device__ __forceinline__ unsigned lds_offset(const void* p) {
  return (unsigned)(unsigned long)(__attribute__((address_space(3))) const void*)p;
}
// 2D bf16 tile load via Tensor Data Mover (D# per CDNA5 ISA).
// pad_int_code: pad after 2^(c+1) dwords; pad_amt_code: pad (c+1) dwords.
static __device__ __forceinline__ void tdm_load_2d_bf16(
    unsigned lds_addr, const void* gaddr,
    unsigned tile_d0, unsigned tile_d1,
    unsigned tensor_d0, unsigned tensor_d1,
    unsigned long stride0,
    unsigned pad_int_code, unsigned pad_amt_code) {
  unsigned long ga = (unsigned long)gaddr;
  u32x4 g0;
  g0[0] = 1u;                                        // count=1 (valid user D#)
  g0[1] = lds_addr;                                  // lds_addr (bytes)
  g0[2] = (unsigned)(ga & 0xffffffffu);              // global_addr[31:0]
  g0[3] = (unsigned)((ga >> 32) & 0x01ffffffu)       // global_addr[56:32]
        | (2u << 30);                                // type=2 ("image")
  i32x8 g1;
  g1[0] = (int)((1u << 16)                           // data_size: 2 bytes
        | (1u << 20)                                 // pad_enable
        | (pad_int_code << 22) | (pad_amt_code << 25));
  g1[1] = (int)((tensor_d0 & 0xffffu) << 16);        // [47:32]=abar=0, [63:48]=td0.lo
  g1[2] = (int)((tensor_d0 >> 16) | ((tensor_d1 & 0xffffu) << 16));
  g1[3] = (int)((tensor_d1 >> 16) | (tile_d0 << 16));
  g1[4] = (int)(tile_d1 & 0xffffu);                  // tile_dim2 = 0
  g1[5] = (int)(unsigned)(stride0 & 0xffffffffu);    // tensor_dim0_stride[31:0]
  g1[6] = (int)(unsigned)((stride0 >> 32) & 0xffffu);// stride0[47:32], stride1=0
  g1[7] = 0;
  i32x4 z4 = {0, 0, 0, 0};
#if defined(__clang_major__) && (__clang_major__ >= 23)
  i32x8 z8 = {0, 0, 0, 0, 0, 0, 0, 0};
  __builtin_amdgcn_tensor_load_to_lds(g0, g1, z4, z4, z8, 0);
#else
  __builtin_amdgcn_tensor_load_to_lds(g0, g1, z4, z4, 0);
#endif
}
#endif

// ---------------- 1. X fp32 -> bf16 ----------------
__global__ void __launch_bounds__(256) k_convert_x(const float* __restrict__ X,
                                                   __bf16* __restrict__ Xb) {
  long i = ((long)blockIdx.x * 256 + threadIdx.x) * 4;
  v4f x = *(const v4f*)(X + i);
  v4bf y;
#pragma unroll
  for (int j = 0; j < 4; j++) y[j] = (__bf16)x[j];
  *(v4bf*)(Xb + i) = y;
}

// ---------------- 2. fused QKV projection: Xb @ [Wq|Wk|Wv] ----------------
// grid (96, 2), block 256 (8 waves). 128x128 tile per block, K-step 32.
// A tile staged by the Tensor Data Mover; B tile fp32->bf16 with packed b32 DS stores.
__global__ void __launch_bounds__(256) k_qkv(const __bf16* __restrict__ Xb,
                                             const float* __restrict__ Wq,
                                             const float* __restrict__ Wk,
                                             const float* __restrict__ Wv,
                                             __bf16* __restrict__ Qb,
                                             __bf16* __restrict__ Kb,
                                             __bf16* __restrict__ Vt) {
  __shared__ __bf16 As[128 * LDSTR];
  __shared__ __bf16 Bs[128 * LDSTR];
  const int bx = blockIdx.x, by = blockIdx.y;
  const int matsel = bx >> 5;                       // 0:q 1:k 2:v
  const float* Wm = (matsel == 0) ? Wq : (matsel == 1) ? Wk : Wv;
  const int nbase = (bx & 31) * 128;
  const int tid = threadIdx.x;
  const int wave = tid >> 5, lane = tid & 31;
  const int mb = (wave >> 2) * 64, nb = (wave & 3) * 32;
  const int fr = lane & 15, kb8 = (lane >> 4) * 8;

  v8f acc[4][2];
#pragma unroll
  for (int mt = 0; mt < 4; mt++)
#pragma unroll
    for (int nt = 0; nt < 2; nt++) acc[mt][nt] = (v8f)0.0f;

  // B staging: thread -> (2 k-rows) x (8 n); lanes 0..15 sweep n (coalesced)
  const int bn0 = (tid & 15) * 8, bk2 = (tid >> 4) * 2;
#if !USE_TDM
  const int ar = tid >> 1, ak = (tid & 1) * 16;     // manual A staging fallback
#endif

  for (int k0 = 0; k0 < N_; k0 += 32) {
#if USE_TDM
    if (wave == 0)
      tdm_load_2d_bf16(lds_offset(&As[0]), Xb + (long)(by * 128) * N_ + k0,
                       /*tile*/ 32, 128, /*tensor*/ N_, M_, /*stride0*/ N_,
                       /*pad: every 16 dwords add 4 dwords*/ 3, 3);
#else
    {
      const __bf16* ga = Xb + (long)(by * 128 + ar) * N_ + k0 + ak;
      *(v8bf*)(&As[ar * LDSTR + ak])     = *(const v8bf*)ga;
      *(v8bf*)(&As[ar * LDSTR + ak + 8]) = *(const v8bf*)(ga + 8);
    }
#endif
    // stage B tile fp32 -> bf16, transposed into [n][k], packed pair stores
    const float* gb = Wm + (long)(k0 + bk2) * N_ + nbase + bn0;
    v4f r0a = *(const v4f*)(gb);
    v4f r0b = *(const v4f*)(gb + 4);
    v4f r1a = *(const v4f*)(gb + N_);
    v4f r1b = *(const v4f*)(gb + N_ + 4);
#pragma unroll
    for (int i = 0; i < 4; i++) {
      v2bf p0; p0[0] = (__bf16)r0a[i]; p0[1] = (__bf16)r1a[i];
      *(v2bf*)&Bs[(bn0 + i) * LDSTR + bk2] = p0;
      v2bf p1; p1[0] = (__bf16)r0b[i]; p1[1] = (__bf16)r1b[i];
      *(v2bf*)&Bs[(bn0 + 4 + i) * LDSTR + bk2] = p1;
    }
    if (k0 + 32 < N_) __builtin_prefetch(gb + 32L * N_, 0, 0);  // global_prefetch_b8
#if USE_TDM
    if (wave == 0) __builtin_amdgcn_s_wait_tensorcnt((short)0);
#endif
    __syncthreads();

    v16bf af[4], bfr[2];
#pragma unroll
    for (int mt = 0; mt < 4; mt++) {
      const __bf16* p = &As[(mb + mt * 16 + fr) * LDSTR + kb8];
      af[mt] = cat16(*(const v8bf*)p, *(const v8bf*)(p + 16));
    }
#pragma unroll
    for (int nt = 0; nt < 2; nt++) {
      const __bf16* p = &Bs[(nb + nt * 16 + fr) * LDSTR + kb8];
      bfr[nt] = cat16(*(const v8bf*)p, *(const v8bf*)(p + 16));
    }
#pragma unroll
    for (int mt = 0; mt < 4; mt++)
#pragma unroll
      for (int nt = 0; nt < 2; nt++)
        acc[mt][nt] = wmma_bf16(af[mt], bfr[nt], acc[mt][nt]);
    __syncthreads();
  }

  const int fm = (lane >> 4) * 8;
#pragma unroll
  for (int mt = 0; mt < 4; mt++)
#pragma unroll
    for (int nt = 0; nt < 2; nt++)
#pragma unroll
      for (int i = 0; i < 8; i++) {
        int m  = by * 128 + mb + mt * 16 + fm + i;
        int cc = nbase + nb + nt * 16 + fr;
        __bf16 v = (__bf16)acc[mt][nt][i];
        if (matsel == 0) {
          Qb[(long)m * N_ + cc] = v;
        } else {
          int h = cc >> 7, d = cc & 127;
          if (matsel == 1) Kb[((long)h * T_ + P_ + m) * D_ + d] = v;
          else             Vt[((long)h * D_ + d) * T_ + P_ + m] = v;
        }
      }
}

// ---------------- 3. cache_K fp32 -> Kb bf16 (t < P) ----------------
__global__ void __launch_bounds__(256) k_convert_k(const float* __restrict__ cacheK,
                                                   __bf16* __restrict__ Kb) {
  long i = ((long)blockIdx.x * 256 + threadIdx.x) * 4;   // over H*P*D
  int  h = (int)(i / ((long)P_ * D_));
  long rem = i - (long)h * P_ * D_;
  int  t = (int)(rem / D_);
  int  d = (int)(rem & (D_ - 1));
  long off = ((long)h * T_ + t) * D_ + d;
  v4f x = *(const v4f*)(cacheK + off);
  v4bf y;
#pragma unroll
  for (int j = 0; j < 4; j++) y[j] = (__bf16)x[j];
  *(v4bf*)(Kb + off) = y;
}

// ---------------- 4. cache_V fp32 -> Vt[h][d][t] bf16 (tiled transpose) ----------------
__global__ void __launch_bounds__(256) k_transpose_v(const float* __restrict__ cacheV,
                                                     __bf16* __restrict__ Vt) {
  __shared__ float tile[32][33];
  const int h = blockIdx.z, tb = blockIdx.y * 32, db = blockIdx.x * 32;
  const int tid = threadIdx.x;
  {
    int lt = tid >> 3, ld4 = (tid & 7) * 4;
    v4f x = *(const v4f*)(cacheV + ((long)h * T_ + tb + lt) * D_ + db + ld4);
#pragma unroll
    for (int j = 0; j < 4; j++) tile[lt][ld4 + j] = x[j];
  }
  __syncthreads();
  {
    int ld = tid >> 3, lt4 = (tid & 7) * 4;
    v4bf y;
#pragma unroll
    for (int j = 0; j < 4; j++) y[j] = (__bf16)tile[lt4 + j][ld];
    *(v4bf*)(Vt + ((long)h * D_ + db + ld) * T_ + tb + lt4) = y;
  }
}

// ---------------- 5. scores = Qh @ Kh^T -> weights region (pre-softmax) ----------------
// grid (T/64, M/128, H), block 256. K-tile (64 t x 128 d bf16) staged once via TDM,
// removing the 8x redundant global fetch (one per wave).
__global__ void __launch_bounds__(256) k_scores(const __bf16* __restrict__ Qb,
                                                const __bf16* __restrict__ Kb,
                                                float* __restrict__ Sc) {
  __shared__ __bf16 Ks[64 * KSTR];                  // 272B rows (16B aligned)
  const int h  = blockIdx.z;
  const int t0 = blockIdx.x * 64;
  const int m0 = blockIdx.y * 128 + (threadIdx.x >> 5) * 16;
  const int wave = threadIdx.x >> 5;
  const int lane = threadIdx.x & 31;
  const int fr = lane & 15, kb8 = (lane >> 4) * 8;

#if USE_TDM
  if (wave == 0) {
    tdm_load_2d_bf16(lds_offset(&Ks[0]), Kb + ((long)h * T_ + t0) * D_,
                     /*tile*/ D_, 64, /*tensor*/ D_, T_, /*stride0*/ D_,
                     /*pad: every 64 dwords add 4 dwords*/ 5, 3);
    __builtin_amdgcn_s_wait_tensorcnt((short)0);
  }
#else
  {
    int r = threadIdx.x >> 2, c = (threadIdx.x & 3) * 32;
    const __bf16* src = Kb + ((long)h * T_ + t0 + r) * D_ + c;
#pragma unroll
    for (int j = 0; j < 4; j++)
      *(v8bf*)(&Ks[r * KSTR + c + j * 8]) = *(const v8bf*)(src + j * 8);
  }
#endif
  __syncthreads();

  v8f acc[4];
#pragma unroll
  for (int nt = 0; nt < 4; nt++) acc[nt] = (v8f)0.0f;

  const __bf16* qrow = Qb + (long)(m0 + fr) * N_ + h * D_ + kb8;
#pragma unroll
  for (int k0 = 0; k0 < D_; k0 += 32) {
    v16bf a = cat16(*(const v8bf*)(qrow + k0), *(const v8bf*)(qrow + k0 + 16));
#pragma unroll
    for (int nt = 0; nt < 4; nt++) {
      const __bf16* kr = &Ks[(nt * 16 + fr) * KSTR + k0 + kb8];
      v16bf b = cat16(*(const v8bf*)kr, *(const v8bf*)(kr + 16));
      acc[nt] = wmma_bf16(a, b, acc[nt]);
    }
  }
  const int fm = (lane >> 4) * 8;
#pragma unroll
  for (int nt = 0; nt < 4; nt++)
#pragma unroll
    for (int i = 0; i < 8; i++)
      Sc[((long)h * M_ + m0 + fm + i) * T_ + t0 + nt * 16 + fr] = acc[nt][i];
}

// ---------------- 6. dual softmax (clean + noise/1.5) ----------------
static __device__ __forceinline__ float block_reduce(float v, float* red, bool isMax) {
  int tid = threadIdx.x;
  red[tid] = v;
  __syncthreads();
#pragma unroll
  for (int s = 128; s > 0; s >>= 1) {
    if (tid < s) red[tid] = isMax ? fmaxf(red[tid], red[tid + s]) : red[tid] + red[tid + s];
    __syncthreads();
  }
  float r = red[0];
  __syncthreads();
  return r;
}

__global__ void __launch_bounds__(256) k_softmax(const float* __restrict__ noise,
                                                 float* __restrict__ Wt,
                                                 float* __restrict__ Pt) {
  __shared__ float red[256];
  const long base = (long)blockIdx.x * T_;       // blockIdx.x == h*M + m
  const int tid = threadIdx.x;
  float s[17], z[17];
  float mx1 = -3.0e38f, mx2 = -3.0e38f;
#pragma unroll
  for (int j = 0; j < 17; j++) {
    int idx = tid + j * 256;
    float sv = Wt[base + idx];
    float nz = noise[base + idx];
    s[j] = sv;
    z[j] = (sv + nz) * (1.0f / 1.5f);
    mx1 = fmaxf(mx1, s[j]);
    mx2 = fmaxf(mx2, z[j]);
  }
  mx1 = block_reduce(mx1, red, true);
  mx2 = block_reduce(mx2, red, true);
  float sm1 = 0.0f, sm2 = 0.0f;
#pragma unroll
  for (int j = 0; j < 17; j++) {
    s[j] = __expf(s[j] - mx1); sm1 += s[j];
    z[j] = __expf(z[j] - mx2); sm2 += z[j];
  }
  sm1 = block_reduce(sm1, red, false);
  sm2 = block_reduce(sm2, red, false);
  float r1 = 1.0f / sm1, r2 = 1.0f / sm2;
#pragma unroll
  for (int j = 0; j < 17; j++) {
    int idx = tid + j * 256;
    Wt[base + idx] = s[j] * r1;
    Pt[base + idx] = z[j] * r2;
  }
}

// ---------------- 7. out = weights @ V -> [H,M,D] flat ----------------
__global__ void __launch_bounds__(256) k_out(const float* __restrict__ Wt,
                                             const __bf16* __restrict__ Vt,
                                             float* __restrict__ Out) {
  const int h  = blockIdx.z;
  const int d0 = blockIdx.x * 64;
  const int m0 = blockIdx.y * 128 + (threadIdx.x >> 5) * 16;
  const int lane = threadIdx.x & 31;
  const int fr = lane & 15, kb8 = (lane >> 4) * 8;
  v8f acc[4];
#pragma unroll
  for (int nt = 0; nt < 4; nt++) acc[nt] = (v8f)0.0f;

  const float* wrow = Wt + ((long)h * M_ + m0 + fr) * T_ + kb8;
  for (int k0 = 0; k0 < T_; k0 += 32) {
    v4f w0 = *(const v4f*)(wrow + k0);
    v4f w1 = *(const v4f*)(wrow + k0 + 4);
    v4f w2 = *(const v4f*)(wrow + k0 + 16);
    v4f w3 = *(const v4f*)(wrow + k0 + 20);
    v16bf a;
#pragma unroll
    for (int i = 0; i < 4; i++) {
      a[i]      = (__bf16)w0[i];
      a[4 + i]  = (__bf16)w1[i];
      a[8 + i]  = (__bf16)w2[i];
      a[12 + i] = (__bf16)w3[i];
    }
    if (k0 + 32 < T_) __builtin_prefetch(wrow + k0 + 32, 0, 0);
#pragma unroll
    for (int nt = 0; nt < 4; nt++) {
      const __bf16* vr = Vt + ((long)h * D_ + d0 + nt * 16 + fr) * T_ + k0 + kb8;
      v16bf b = cat16(*(const v8bf*)vr, *(const v8bf*)(vr + 16));
      acc[nt] = wmma_bf16(a, b, acc[nt]);
    }
  }
  const int fm = (lane >> 4) * 8;
#pragma unroll
  for (int nt = 0; nt < 4; nt++)
#pragma unroll
    for (int i = 0; i < 8; i++)
      Out[(long)h * M_ * D_ + (long)(m0 + fm + i) * D_ + d0 + nt * 16 + fr] = acc[nt][i];
}

// ---------------- launcher ----------------
extern "C" void kernel_launch(void* const* d_in, const int* in_sizes, int n_in,
                              void* d_out, int out_size, void* d_ws, size_t ws_size,
                              hipStream_t stream) {
  (void)in_sizes; (void)n_in; (void)out_size; (void)ws_size;
  const float* X      = (const float*)d_in[0];
  const float* Wq     = (const float*)d_in[1];
  const float* Wk     = (const float*)d_in[2];
  const float* Wv     = (const float*)d_in[3];
  const float* noise  = (const float*)d_in[4];
  const float* cacheK = (const float*)d_in[5];
  const float* cacheV = (const float*)d_in[6];

  float* out     = (float*)d_out;                       // [H,M,D] flat == [M,H*D]
  float* weights = out + (long)M_ * N_;                 // [H,M,T]
  float* perturb = weights + (long)H_ * M_ * T_;        // [H,M,T]

  char* ws = (char*)d_ws;
  const long SZ_XB = (long)M_ * N_ * 2;                 // 2 MB
  const long SZ_KB = (long)H_ * T_ * D_ * 2;            // 35.65 MB
  __bf16* Xb = (__bf16*)(ws);
  __bf16* Qb = (__bf16*)(ws + SZ_XB);
  __bf16* Kb = (__bf16*)(ws + 2 * SZ_XB);
  __bf16* Vt = (__bf16*)(ws + 2 * SZ_XB + SZ_KB);       // total ~76 MB

  k_convert_x  <<<dim3((M_ * N_) / 1024), 256, 0, stream>>>(X, Xb);
  k_convert_k  <<<dim3((H_ * P_ * D_) / 1024), 256, 0, stream>>>(cacheK, Kb);
  k_transpose_v<<<dim3(D_ / 32, P_ / 32, H_), 256, 0, stream>>>(cacheV, Vt);
  k_qkv        <<<dim3(96, 2), 256, 0, stream>>>(Xb, Wq, Wk, Wv, Qb, Kb, Vt);
  k_scores     <<<dim3(T_ / 64, M_ / 128, H_), 256, 0, stream>>>(Qb, Kb, weights);
  k_softmax    <<<dim3(H_ * M_), 256, 0, stream>>>(noise, weights, perturb);
  k_out        <<<dim3(D_ / 64, M_ / 128, H_), 256, 0, stream>>>(weights, Vt, out);
}
